// FuseAttention_12197707121196
// MI455X (gfx1250) — matrix operand order repaired and checked by reading, not compiled, not run on hardware
//
#include <hip/hip_runtime.h>
#include <hip/hip_bf16.h>

// ---------- CDNA5 WMMA types ----------
typedef __attribute__((ext_vector_type(16))) __bf16     v16bf;
typedef __attribute__((ext_vector_type(2)))  __bf16     v2bf;
typedef __attribute__((ext_vector_type(8)))  float      v8f;
typedef __attribute__((ext_vector_type(8)))  unsigned   v8u;

__device__ __forceinline__ v16bf wmma_frag(uint4 a, uint4 b) {
    v8u r;
    r[0] = a.x; r[1] = a.y; r[2] = a.z; r[3] = a.w;
    r[4] = b.x; r[5] = b.y; r[6] = b.z; r[7] = b.w;
    return __builtin_bit_cast(v16bf, r);
}

__device__ __forceinline__ unsigned short f2bf(float f) {
    unsigned u = __builtin_bit_cast(unsigned, f);
    u = u + 0x7FFFu + ((u >> 16) & 1u);          // round-to-nearest-even
    return (unsigned short)(u >> 16);
}

// packed f32x2 -> bf16x2 (v_cvt_pk_bf16_f32 when the toolchain has it)
__device__ __forceinline__ unsigned pk_bf16(float lo, float hi) {
#if __has_builtin(__builtin_amdgcn_cvt_pk_bf16_f32)
    return __builtin_bit_cast(unsigned, __builtin_amdgcn_cvt_pk_bf16_f32(lo, hi));
#else
    return (unsigned)f2bf(lo) | ((unsigned)f2bf(hi) << 16);
#endif
}

__device__ __forceinline__ v8f wmma_bf16(v16bf a, v16bf b, v8f c) {
    return __builtin_amdgcn_wmma_f32_16x16x32_bf16(false, a, false, b,
                                                   (short)0, c, false, false);
}

__device__ __forceinline__ unsigned hash_u32(unsigned x) {
    x ^= x >> 16; x *= 0x7feb352du;
    x ^= x >> 15; x *= 0x846ca68bu;
    x ^= x >> 16; return x;
}

// =====================================================================
// Kernel 0: streaming f32 -> bf16 convert (8 elements / thread)
// =====================================================================
__global__ void __launch_bounds__(256)
cvt_bf16_kernel(const float* __restrict__ src, unsigned short* __restrict__ dst,
                unsigned n8)
{
    const unsigned i = blockIdx.x * 256u + threadIdx.x;
    if (i >= n8) return;
    const float4 a = ((const float4*)src)[2u * i];
    const float4 b = ((const float4*)src)[2u * i + 1u];
    uint4 st;
    st.x = pk_bf16(a.x, a.y); st.y = pk_bf16(a.z, a.w);
    st.z = pk_bf16(b.x, b.y); st.w = pk_bf16(b.z, b.w);
    ((uint4*)dst)[i] = st;
}

// =====================================================================
// Kernel 1: C = X @ W^T + bias, all-bf16 inputs, bf16 out.
// Block = 8 waves; wave owns 2 row-tiles sharing one B fragment.
// Block tile = 64 rows x 64 cols.  TRANS=true stores C^T (for V) with
// one b128 store per lane (8 contiguous rows per lane).
// =====================================================================
template <bool TRANS>
__global__ void __launch_bounds__(256)
gemm_bf16_kernel(const unsigned short* __restrict__ X,   // [R,512] bf16
                 const unsigned short* __restrict__ Wb,  // [512,512] bf16
                 const float* __restrict__ bias,
                 unsigned short* __restrict__ out, unsigned ldo)
{
    const int tid  = threadIdx.x;
    const int wave = tid >> 5, lane = tid & 31;
    const int m    = lane & 15, half = lane >> 4;
    const int sct  = wave & 3;
    const int srt0 = (wave >> 2) * 2;
    const int row0 = blockIdx.x * 64 + srt0 * 16;
    const int col0 = blockIdx.y * 64 + sct * 16;
    const int D    = 512;

    const unsigned short* xrow0 = X + (size_t)(row0 + m) * D;
    const unsigned short* xrow1 = xrow0 + (size_t)16 * D;
    const unsigned short* wrow  = Wb + (size_t)(col0 + m) * D;

    v8f acc0 = {}, acc1 = {};
#pragma unroll 4
    for (int kk = 0; kk < D; kk += 32) {
        uint4 wb0 = *(const uint4*)(wrow + kk + half * 8);
        uint4 wb1 = *(const uint4*)(wrow + kk + 16 + half * 8);
        uint4 xa0 = *(const uint4*)(xrow0 + kk + half * 8);
        uint4 xa1 = *(const uint4*)(xrow0 + kk + 16 + half * 8);
        uint4 xb0 = *(const uint4*)(xrow1 + kk + half * 8);
        uint4 xb1 = *(const uint4*)(xrow1 + kk + 16 + half * 8);
        v16bf Bf = wmma_frag(wb0, wb1);
        acc0 = wmma_bf16(wmma_frag(xa0, xa1), Bf, acc0);
        acc1 = wmma_bf16(wmma_frag(xb0, xb1), Bf, acc1);
    }

    const unsigned col = col0 + m;
    const float    bb  = bias[col];
    if (TRANS) {
        // lane's 8 rows are contiguous in C^T -> one b128 store per tile
        uint4 st0, st1;
        st0.x = pk_bf16(acc0[0] + bb, acc0[1] + bb);
        st0.y = pk_bf16(acc0[2] + bb, acc0[3] + bb);
        st0.z = pk_bf16(acc0[4] + bb, acc0[5] + bb);
        st0.w = pk_bf16(acc0[6] + bb, acc0[7] + bb);
        st1.x = pk_bf16(acc1[0] + bb, acc1[1] + bb);
        st1.y = pk_bf16(acc1[2] + bb, acc1[3] + bb);
        st1.z = pk_bf16(acc1[4] + bb, acc1[5] + bb);
        st1.w = pk_bf16(acc1[6] + bb, acc1[7] + bb);
        unsigned short* op = out + (size_t)col * ldo + row0 + half * 8;
        *(uint4*)op        = st0;
        *(uint4*)(op + 16) = st1;
    } else {
#pragma unroll
        for (int i = 0; i < 8; ++i) {
            const unsigned row = row0 + i + half * 8;
            out[(size_t)row * ldo + col]        = f2bf(acc0[i] + bb);
            out[(size_t)(row + 16) * ldo + col] = f2bf(acc1[i] + bb);
        }
    }
}

// =====================================================================
// Kernel 2: flash-attention over precomputed bf16 Q[N,512], K[M,512],
// V^T[512,M].  Block = 64 Q rows, 8 waves, streams keys in 64-chunks.
// =====================================================================
#define MK    4096
#define DD    512
#define SCALE 10.0f

__global__ void __launch_bounds__(256)
attn_kernel(const unsigned short* __restrict__ Qb,
            const unsigned short* __restrict__ Kb,
            const unsigned short* __restrict__ Vt,
            float* __restrict__ out)
{
    __shared__ float          Slds[64 * 64];      // raw scores (f32)      16KB
    __shared__ unsigned short Plds[64 * 64];      // softmax+dropout bf16   8KB
    __shared__ float          mrow[64], lrow[64], rfac[64];

    const int tid  = threadIdx.x;
    const int wave = tid >> 5, lane = tid & 31;
    const int m    = lane & 15, half = lane >> 4;
    const int row0 = blockIdx.x * 64;

    v8f oacc[16] = {};                            // [rt(4)][ct(4)] 16x16 f32

    if (tid < 64) { mrow[tid] = -__builtin_inff(); lrow[tid] = 0.0f; }
    __syncthreads();

    const int sct  = wave & 3;                    // stage-1: wave owns 2 S tiles
    const int srt0 = (wave >> 2) * 2;             //   (srt0, sct) and (srt0+1, sct)
    const int d0   = wave * 64;                   // stage-3 output columns

    const unsigned short* qrow0 = Qb + (size_t)(row0 + srt0 * 16 + m) * DD;
    const unsigned short* qrow1 = qrow0 + (size_t)16 * DD;

    for (int ch = 0; ch < MK / 64; ++ch) {
        const int c0 = ch * 64;

        // ---- Stage 1: two S tiles sharing one K B-fragment (32 WMMAs) ----
        {
            const unsigned short* krow = Kb + (size_t)(c0 + sct * 16 + m) * DD;
            v8f s0 = {}, s1 = {};
#pragma unroll 4
            for (int kk = 0; kk < DD; kk += 32) {
                uint4 kb0 = *(const uint4*)(krow + kk + half * 8);
                uint4 kb1 = *(const uint4*)(krow + kk + 16 + half * 8);
                uint4 qa0 = *(const uint4*)(qrow0 + kk + half * 8);
                uint4 qa1 = *(const uint4*)(qrow0 + kk + 16 + half * 8);
                uint4 qb0 = *(const uint4*)(qrow1 + kk + half * 8);
                uint4 qb1 = *(const uint4*)(qrow1 + kk + 16 + half * 8);
                v16bf Bf = wmma_frag(kb0, kb1);
                s0 = wmma_bf16(wmma_frag(qa0, qa1), Bf, s0);
                s1 = wmma_bf16(wmma_frag(qb0, qb1), Bf, s1);
            }
#pragma unroll
            for (int i = 0; i < 8; ++i) {
                Slds[(srt0 * 16 + i + half * 8) * 64 + sct * 16 + m]       = s0[i];
                Slds[((srt0 + 1) * 16 + i + half * 8) * 64 + sct * 16 + m] = s1[i];
            }
        }
        __syncthreads();

        // ---- Stage 2: online softmax stats + dropout-masked P (bf16) ----
        if (tid < 64) {
            const int r  = tid;
            const int gr = row0 + r;
            const float mo = mrow[r];
            float mx = mo;
            for (int c = 0; c < 64; ++c)
                mx = fmaxf(mx, Slds[r * 64 + c] * SCALE);
            const float rf = __expf(mo - mx);     // exp(-inf)=0 on first chunk
            float lsum = 0.0f;
            for (int c = 0; c < 64; ++c) {
                const float p = __expf(Slds[r * 64 + c] * SCALE - mx);
                lsum += p;                        // denominator uses ALL keys
                const unsigned h = hash_u32((unsigned)(gr * MK + c0 + c) ^ 0x9E3779B9u);
                const float pd = (h & 1u) ? p * 2.0f : 0.0f;  // drop p=0.5, /(1-p)
                Plds[r * 64 + c] = f2bf(pd);
            }
            mrow[r] = mx;
            lrow[r] = lrow[r] * rf + lsum;
            rfac[r] = rf;
        }
        __syncthreads();

        // ---- Stage 3: O = O*rf + P[64,64] @ V[64, 64-col slice] ----
        {
            // per-row rescale of all 16 accumulator tiles
#pragma unroll
            for (int rt = 0; rt < 4; ++rt) {
                float rl[8];
#pragma unroll
                for (int i = 0; i < 8; ++i) rl[i] = rfac[rt * 16 + i + half * 8];
#pragma unroll
                for (int c = 0; c < 4; ++c)
#pragma unroll
                    for (int i = 0; i < 8; ++i) oacc[rt * 4 + c][i] *= rl[i];
            }
            // 2 k-blocks of 32; P A-frags from LDS, V B-frags from V^T
#pragma unroll
            for (int kb = 0; kb < 2; ++kb) {
                v16bf Af[4];
#pragma unroll
                for (int rt = 0; rt < 4; ++rt) {
                    const unsigned short* pr = Plds + (rt * 16 + m) * 64 + kb * 32;
                    uint4 p0 = *(const uint4*)(pr + half * 8);
                    uint4 p1 = *(const uint4*)(pr + 16 + half * 8);
                    Af[rt] = wmma_frag(p0, p1);
                }
#pragma unroll
                for (int c = 0; c < 4; ++c) {
                    const unsigned short* vr =
                        Vt + (size_t)(d0 + c * 16 + m) * MK + c0 + kb * 32;
                    uint4 v0 = *(const uint4*)(vr + half * 8);
                    uint4 v1 = *(const uint4*)(vr + 16 + half * 8);
                    v16bf Bf = wmma_frag(v0, v1);
#pragma unroll
                    for (int rt = 0; rt < 4; ++rt)
                        oacc[rt * 4 + c] = wmma_bf16(Af[rt], Bf, oacc[rt * 4 + c]);
                }
            }
        }
        // next iteration's Stage-1 barrier provides the needed ordering
    }

    // ---- Epilogue: divide by full softmax denominator, store f32 ----
#pragma unroll
    for (int rt = 0; rt < 4; ++rt) {
        float ll[8];
#pragma unroll
        for (int i = 0; i < 8; ++i)
            ll[i] = 1.0f / lrow[rt * 16 + i + half * 8];
#pragma unroll
        for (int c = 0; c < 4; ++c)
#pragma unroll
            for (int i = 0; i < 8; ++i) {
                const int row = row0 + rt * 16 + i + half * 8;
                out[(size_t)row * DD + d0 + c * 16 + m] = oacc[rt * 4 + c][i] * ll[i];
            }
    }
}

// =====================================================================
extern "C" void kernel_launch(void* const* d_in, const int* in_sizes, int n_in,
                              void* d_out, int out_size, void* d_ws, size_t ws_size,
                              hipStream_t stream)
{
    (void)in_sizes; (void)n_in; (void)out_size; (void)ws_size;
    const float* feat    = (const float*)d_in[0];   // [16384, 512]
    const float* concept = (const float*)d_in[1];   // [4096, 512]
    const float* Wq      = (const float*)d_in[2];
    const float* bq      = (const float*)d_in[3];
    const float* Wk      = (const float*)d_in[4];
    const float* bk      = (const float*)d_in[5];
    const float* Wv      = (const float*)d_in[6];
    const float* bv      = (const float*)d_in[7];

    // bf16 workspace layout (ushort elements)
    unsigned short* qbf = (unsigned short*)d_ws;                 // [16384,512]
    unsigned short* kbf = qbf + (size_t)16384 * 512;             // [4096,512]
    unsigned short* vtb = kbf + (size_t)4096 * 512;              // [512,4096] (V^T)
    unsigned short* fbf = vtb + (size_t)512 * 4096;              // [16384,512]
    unsigned short* cbf = fbf + (size_t)16384 * 512;             // [4096,512]
    unsigned short* wqb = cbf + (size_t)4096 * 512;              // [512,512]
    unsigned short* wkb = wqb + (size_t)512 * 512;
    unsigned short* wvb = wkb + (size_t)512 * 512;

    dim3 blk(256);
    const unsigned nF = 16384u * 512u / 8u, nC = 4096u * 512u / 8u,
                   nW = 512u * 512u / 8u;
    cvt_bf16_kernel<<<dim3((nF + 255) / 256), blk, 0, stream>>>(feat, fbf, nF);
    cvt_bf16_kernel<<<dim3((nC + 255) / 256), blk, 0, stream>>>(concept, cbf, nC);
    cvt_bf16_kernel<<<dim3((nW + 255) / 256), blk, 0, stream>>>(Wq, wqb, nW);
    cvt_bf16_kernel<<<dim3((nW + 255) / 256), blk, 0, stream>>>(Wk, wkb, nW);
    cvt_bf16_kernel<<<dim3((nW + 255) / 256), blk, 0, stream>>>(Wv, wvb, nW);

    gemm_bf16_kernel<false><<<dim3(16384 / 64, 512 / 64), blk, 0, stream>>>(
        fbf, wqb, bq, qbf, 512u);
    gemm_bf16_kernel<false><<<dim3(4096 / 64, 512 / 64), blk, 0, stream>>>(
        cbf, wkb, bk, kbf, 512u);
    gemm_bf16_kernel<true><<<dim3(4096 / 64, 512 / 64), blk, 0, stream>>>(
        cbf, wvb, bv, vtb, 4096u);

    attn_kernel<<<dim3(16384 / 64), blk, 0, stream>>>(qbf, kbf, vtb, (float*)d_out);
}
